// GCNModel_39350490366682
// MI455X (gfx1250) — compile-verified, matrix operand order checked by reading, not run on hardware
//
#include <hip/hip_runtime.h>
#include <hip/hip_bf16.h>
#include <math.h>

typedef __attribute__((ext_vector_type(16))) _Float16 v16h;
typedef __attribute__((ext_vector_type(8)))  _Float16 v8h;
typedef __attribute__((ext_vector_type(8)))  float    v8f;

#define NUM_NODES_PER_GRAPH 111
#define AGG_DIM 669
#define BN_EPS 1e-5f

// ---------------------------------------------------------------------------
// Degree / normalization
// ---------------------------------------------------------------------------
__global__ void deg_init_kernel(float* deg, int n) {
    int i = blockIdx.x * blockDim.x + threadIdx.x;
    if (i < n) deg[i] = 1.0f;               // self-loop contributes 1
}

__global__ void deg_edges_kernel(const int* __restrict__ edge_index, float* deg, int E) {
    int e = blockIdx.x * blockDim.x + threadIdx.x;
    if (e < E) atomicAdd(&deg[edge_index[E + e]], 1.0f);
}

__global__ void deg_rsqrt_kernel(float* deg, int n) {
    int i = blockIdx.x * blockDim.x + threadIdx.x;
    if (i < n) deg[i] = rsqrtf(deg[i]);     // deg >= 1 always (self loop)
}

// ---------------------------------------------------------------------------
// Packing kernels: f32 -> f16, zero-padded, fragment-friendly layouts.
//   pack_a: Xh[Mpad, Kpad] = X[M, K]        (row-major, zero pad)
//   pack_b: Wt[Nout, Kpad] = W[K, Nout]^T   (transposed, zero pad)
// Kpad is a power of two (128/64/32) -> logKpad shifts.
// ---------------------------------------------------------------------------
__global__ void pack_a_f16_kernel(const float* __restrict__ X, _Float16* __restrict__ Xh,
                                  int M, int K, int Mpad, int logKpad) {
    int idx = blockIdx.x * blockDim.x + threadIdx.x;
    int total = Mpad << logKpad;
    if (idx < total) {
        int row = idx >> logKpad;
        int k   = idx & ((1 << logKpad) - 1);
        float v = (row < M && k < K) ? X[(size_t)row * K + k] : 0.0f;
        Xh[idx] = (_Float16)v;
    }
}

__global__ void pack_b_f16_kernel(const float* __restrict__ W, _Float16* __restrict__ Wt,
                                  int K, int Nout, int logKpad) {
    int idx = blockIdx.x * blockDim.x + threadIdx.x;
    int total = Nout << logKpad;
    if (idx < total) {
        int n = idx >> logKpad;
        int k = idx & ((1 << logKpad) - 1);
        float v = (k < K) ? W[(size_t)k * Nout + n] : 0.0f;
        Wt[idx] = (_Float16)v;
    }
}

// ---------------------------------------------------------------------------
// WMMA GEMM on packed f16:  Y[Mpad,Nout] = Xh[Mpad,Kpad] @ Wt[Nout,Kpad]^T
// One wave per 16x16 output tile. Fragment layouts per CDNA5 ISA 7.12.2:
//   A (16x32 f16): lane&15 = row; lanes<16 hold K = 0..7 & 16..23,
//                  lanes>=16 hold K = 8..15 & 24..31  -> two contiguous b128s.
//   B (32x16 f16): lane&15 = col; same K striping (Wt is K-contiguous per col).
//   C/D (16x16 f32): lane&15 = col; VGPR r holds row r + 8*(lane>=16).
// No bounds checks anywhere (padded operands), no branches around the WMMA.
// ---------------------------------------------------------------------------
__global__ void gemm_wmma_packed_kernel(const _Float16* __restrict__ Xh,
                                        const _Float16* __restrict__ Wt,
                                        float* __restrict__ Y,
                                        int Kpad, int Nout) {
    const int ntiles = Nout >> 4;
    const int mt = (int)blockIdx.x / ntiles;
    const int nt = (int)blockIdx.x % ntiles;
    const int lane = threadIdx.x & 31;
    const int hsel = lane >> 4;
    const int l15  = lane & 15;
    const int kb   = hsel << 3;             // 0 or 8

    const _Float16* arow = Xh + (size_t)(mt * 16 + l15) * Kpad + kb;
    const _Float16* brow = Wt + (size_t)(nt * 16 + l15) * Kpad + kb;

    v8f c = {};
    for (int k0 = 0; k0 < Kpad; k0 += 32) {
        v8h alo = *(const v8h*)(arow + k0);        // K = k0+kb   .. +7   (b128)
        v8h ahi = *(const v8h*)(arow + k0 + 16);   // K = k0+kb+16 .. +23 (b128)
        v8h blo = *(const v8h*)(brow + k0);
        v8h bhi = *(const v8h*)(brow + k0 + 16);
        v16h a, b;
#pragma unroll
        for (int j = 0; j < 8; ++j) {
            a[j] = alo[j]; a[j + 8] = ahi[j];
            b[j] = blo[j]; b[j + 8] = bhi[j];
        }
        c = __builtin_amdgcn_wmma_f32_16x16x32_f16(
                false, a, false, b, (short)0, c, false, false);
    }

    float* yrow = Y + (size_t)(mt * 16 + (hsel << 3)) * Nout + nt * 16 + l15;
#pragma unroll
    for (int r = 0; r < 8; ++r)
        yrow[(size_t)r * Nout] = c[r];
}

// ---------------------------------------------------------------------------
// GCN propagation (float4-vectorized): self-loop init, edge atomic scatter,
// bias + ReLU.  F is 64/32/16 -> F/4 groups per row, logFq = log2(F/4).
// ---------------------------------------------------------------------------
__global__ void self_scale4_kernel(const float4* __restrict__ hlin,
                                   const float* __restrict__ dinv,
                                   float4* __restrict__ out, int n, int logFq) {
    int idx = blockIdx.x * blockDim.x + threadIdx.x;
    int total = n << logFq;
    if (idx < total) {
        int node = idx >> logFq;
        float s = dinv[node];
        float sc = s * s;                   // self-loop norm = dinv[i]^2
        float4 v = hlin[idx];
        v.x *= sc; v.y *= sc; v.z *= sc; v.w *= sc;
        out[idx] = v;
    }
}

__global__ void edge_scatter4_kernel(const int* __restrict__ edge_index,
                                     const float4* __restrict__ hlin,
                                     const float* __restrict__ dinv,
                                     float* __restrict__ out, int E, int logFq) {
    int idx = blockIdx.x * blockDim.x + threadIdx.x;
    int total = E << logFq;
    if (idx < total) {
        int e = idx >> logFq;
        int q = idx & ((1 << logFq) - 1);
        int s = edge_index[e];
        int d = edge_index[E + e];
        float norm = dinv[s] * dinv[d];
        float4 v = hlin[((size_t)s << logFq) + q];
        float* o = out + (((size_t)d << logFq) + q) * 4;
        atomicAdd(o + 0, v.x * norm);
        atomicAdd(o + 1, v.y * norm);
        atomicAdd(o + 2, v.z * norm);
        atomicAdd(o + 3, v.w * norm);
    }
}

__global__ void bias_relu4_kernel(float4* __restrict__ out,
                                  const float4* __restrict__ bias, int n, int logFq) {
    int idx = blockIdx.x * blockDim.x + threadIdx.x;
    int total = n << logFq;
    if (idx < total) {
        int q = idx & ((1 << logFq) - 1);
        float4 v = out[idx];
        float4 b = bias[q];
        v.x = fmaxf(v.x + b.x, 0.0f);
        v.y = fmaxf(v.y + b.y, 0.0f);
        v.z = fmaxf(v.z + b.z, 0.0f);
        v.w = fmaxf(v.w + b.w, 0.0f);
        out[idx] = v;
    }
}

// ---------------------------------------------------------------------------
// Multi-aggregation (mean | max | sum) over the 111 contiguous nodes of each
// graph, written at column offset `off` of agg[g, 669].
// ---------------------------------------------------------------------------
__global__ void multi_aggr_kernel(const float* __restrict__ src, int F,
                                  float* __restrict__ agg, int off) {
    int g = blockIdx.x;
    const size_t base = (size_t)g * NUM_NODES_PER_GRAPH * F;
    float* ag = agg + (size_t)g * AGG_DIM + off;
    for (int f = threadIdx.x; f < F; f += blockDim.x) {
        float sm = 0.0f;
        float mx = -INFINITY;
        for (int r = 0; r < NUM_NODES_PER_GRAPH; ++r) {
            float v = src[base + (size_t)r * F + f];
            sm += v;
            mx = fmaxf(mx, v);
        }
        ag[f]         = sm * (1.0f / NUM_NODES_PER_GRAPH);  // mean
        ag[F + f]     = mx;                                  // max
        ag[2 * F + f] = sm;                                  // sum
    }
}

// ---------------------------------------------------------------------------
// Head: BN(eval) -> FC(669->64) -> ReLU -> FC(64->1) -> sigmoid. One block/graph.
// ---------------------------------------------------------------------------
__global__ void head_kernel(const float* __restrict__ agg,
                            const float* __restrict__ gamma,
                            const float* __restrict__ beta,
                            const float* __restrict__ mean,
                            const float* __restrict__ var,
                            const float* __restrict__ fcW,
                            const float* __restrict__ fcb,
                            const float* __restrict__ fc2W,
                            const float* __restrict__ fc2b,
                            float* __restrict__ out) {
    __shared__ float z[AGG_DIM];
    __shared__ float h[64];
    int g = blockIdx.x;
    for (int k = threadIdx.x; k < AGG_DIM; k += blockDim.x) {
        float a = agg[(size_t)g * AGG_DIM + k];
        z[k] = (a - mean[k]) * rsqrtf(var[k] + BN_EPS) * gamma[k] + beta[k];
    }
    __syncthreads();
    int j = threadIdx.x;
    if (j < 64) {
        float acc = fcb[j];
        for (int k = 0; k < AGG_DIM; ++k) acc += z[k] * fcW[k * 64 + j];
        h[j] = fmaxf(acc, 0.0f);
    }
    __syncthreads();
    if (threadIdx.x == 0) {
        float s = fc2b[0];
        for (int jj = 0; jj < 64; ++jj) s += h[jj] * fc2W[jj];
        out[g] = 1.0f / (1.0f + expf(-s));
    }
}

// ---------------------------------------------------------------------------
// Launch
// ---------------------------------------------------------------------------
static inline int cdiv(int a, int b) { return (a + b - 1) / b; }

extern "C" void kernel_launch(void* const* d_in, const int* in_sizes, int n_in,
                              void* d_out, int out_size, void* d_ws, size_t ws_size,
                              hipStream_t stream) {
    const float* x   = (const float*)d_in[0];
    const int*   ei  = (const int*)  d_in[1];
    // d_in[2] = batch (unused: segments are contiguous, 111 nodes/graph)
    const float* W1  = (const float*)d_in[3];
    const float* b1  = (const float*)d_in[4];
    const float* W2  = (const float*)d_in[5];
    const float* b2  = (const float*)d_in[6];
    const float* W3  = (const float*)d_in[7];
    const float* b3  = (const float*)d_in[8];
    const float* bn_gamma = (const float*)d_in[9];
    const float* bn_beta  = (const float*)d_in[10];
    const float* bn_mean  = (const float*)d_in[11];
    const float* bn_var   = (const float*)d_in[12];
    const float* fcW  = (const float*)d_in[13];
    const float* fcb  = (const float*)d_in[14];
    const float* fc2W = (const float*)d_in[15];
    const float* fc2b = (const float*)d_in[16];
    float* out = (float*)d_out;

    const int N  = in_sizes[0] / NUM_NODES_PER_GRAPH;   // 99900 nodes
    const int E  = in_sizes[1] / 2;                      // 1.6M edges
    const int NG = N / NUM_NODES_PER_GRAPH;              // 900 graphs
    const int mtiles = cdiv(N, 16);
    const int Mpad = mtiles * 16;

    // Workspace carving (float units)
    float* ws   = (float*)d_ws;
    float* dinv = ws;                              // N
    float* hlin = dinv + N;                        // Mpad*64 (GEMM out, reused)
    float* h1   = hlin + (size_t)Mpad * 64;        // N*64
    float* h2   = h1   + (size_t)N * 64;           // N*32
    float* h3   = h2   + (size_t)N * 32;           // N*16
    float* agg  = h3   + (size_t)N * 16;           // NG*669
    _Float16* xh  = (_Float16*)(agg + (size_t)NG * AGG_DIM); // Mpad*128 halves (reused)
    _Float16* wth = xh + (size_t)Mpad * 128;                 // up to 64*128 halves

    const int T = 256;

    // --- degrees / dinv (shared by all three conv layers) ---
    deg_init_kernel <<<cdiv(N, T), T, 0, stream>>>(dinv, N);
    deg_edges_kernel<<<cdiv(E, T), T, 0, stream>>>(ei, dinv, E);
    deg_rsqrt_kernel<<<cdiv(N, T), T, 0, stream>>>(dinv, N);

    // --- layer 1: 111 -> 64 (Kpad=128) ---
    pack_a_f16_kernel<<<cdiv(Mpad * 128, T), T, 0, stream>>>(x,  xh,  N, 111, Mpad, 7);
    pack_b_f16_kernel<<<cdiv(64 * 128,  T), T, 0, stream>>>(W1, wth, 111, 64, 7);
    gemm_wmma_packed_kernel<<<mtiles * 4, 32, 0, stream>>>(xh, wth, hlin, 128, 64);
    self_scale4_kernel  <<<cdiv(N * 16, T), T, 0, stream>>>((const float4*)hlin, dinv, (float4*)h1, N, 4);
    edge_scatter4_kernel<<<cdiv(E * 16, T), T, 0, stream>>>(ei, (const float4*)hlin, dinv, h1, E, 4);
    bias_relu4_kernel   <<<cdiv(N * 16, T), T, 0, stream>>>((float4*)h1, (const float4*)b1, N, 4);

    // --- layer 2: 64 -> 32 (Kpad=64) ---
    pack_a_f16_kernel<<<cdiv(Mpad * 64, T), T, 0, stream>>>(h1, xh,  N, 64, Mpad, 6);
    pack_b_f16_kernel<<<cdiv(32 * 64,  T), T, 0, stream>>>(W2, wth, 64, 32, 6);
    gemm_wmma_packed_kernel<<<mtiles * 2, 32, 0, stream>>>(xh, wth, hlin, 64, 32);
    self_scale4_kernel  <<<cdiv(N * 8, T), T, 0, stream>>>((const float4*)hlin, dinv, (float4*)h2, N, 3);
    edge_scatter4_kernel<<<cdiv(E * 8, T), T, 0, stream>>>(ei, (const float4*)hlin, dinv, h2, E, 3);
    bias_relu4_kernel   <<<cdiv(N * 8, T), T, 0, stream>>>((float4*)h2, (const float4*)b2, N, 3);

    // --- layer 3: 32 -> 16 (Kpad=32) ---
    pack_a_f16_kernel<<<cdiv(Mpad * 32, T), T, 0, stream>>>(h2, xh,  N, 32, Mpad, 5);
    pack_b_f16_kernel<<<cdiv(16 * 32,  T), T, 0, stream>>>(W3, wth, 32, 16, 5);
    gemm_wmma_packed_kernel<<<mtiles * 1, 32, 0, stream>>>(xh, wth, hlin, 32, 16);
    self_scale4_kernel  <<<cdiv(N * 4, T), T, 0, stream>>>((const float4*)hlin, dinv, (float4*)h3, N, 2);
    edge_scatter4_kernel<<<cdiv(E * 4, T), T, 0, stream>>>(ei, (const float4*)hlin, dinv, h3, E, 2);
    bias_relu4_kernel   <<<cdiv(N * 4, T), T, 0, stream>>>((float4*)h3, (const float4*)b3, N, 2);

    // --- readout: concat(mean|max|sum) of [x, h1, h2, h3] -> agg[NG, 669] ---
    multi_aggr_kernel<<<NG, 128, 0, stream>>>(x,  111, agg, 0);
    multi_aggr_kernel<<<NG, 128, 0, stream>>>(h1,  64, agg, 333);
    multi_aggr_kernel<<<NG, 128, 0, stream>>>(h2,  32, agg, 525);
    multi_aggr_kernel<<<NG, 128, 0, stream>>>(h3,  16, agg, 621);

    // --- BN + MLP head + sigmoid ---
    head_kernel<<<NG, 64, 0, stream>>>(agg, bn_gamma, bn_beta, bn_mean, bn_var,
                                       fcW, fcb, fc2W, fc2b, out);
}
